// DCN_47339129536901
// MI455X (gfx1250) — compile-verified, matrix operand order checked by reading, not run on hardware
//
#include <hip/hip_runtime.h>

// ---------------------------------------------------------------------------
// Types for CDNA5 WMMA (gfx1250, wave32)
// ---------------------------------------------------------------------------
typedef __attribute__((ext_vector_type(16))) __bf16  v16bf;
typedef __attribute__((ext_vector_type(8)))  float   v8f;
typedef __attribute__((ext_vector_type(4)))  unsigned int uint4v;

__device__ __forceinline__ unsigned short f2bf(float f) {
    unsigned int u = __float_as_uint(f);
    u += 0x7FFFu + ((u >> 16) & 1u);          // round-to-nearest-even
    return (unsigned short)(u >> 16);
}
__device__ __forceinline__ float bf2f(unsigned short h) {
    return __uint_as_float(((unsigned int)h) << 16);
}

// Low 32 bits of a generic pointer to __shared__ = LDS byte address
// (the shared aperture occupies addr[63:32] only; addr[31:0] is the LDS offset).
__device__ __forceinline__ unsigned lds_addr32(const void* p) {
    return (unsigned)(unsigned long long)(uintptr_t)p;
}

// CDNA5 async copy global -> LDS, 16 bytes per lane, tracked by ASYNCcnt.
// GVS mode: 64-bit uniform base in SGPRs + per-lane 32-bit byte offset.
__device__ __forceinline__ void async_load_b128(unsigned lds_dst, unsigned voff,
                                                const void* sbase) {
    asm volatile("global_load_async_to_lds_b128 %0, %1, %2"
                 :: "v"(lds_dst), "v"(voff), "s"(sbase)
                 : "memory");
}
__device__ __forceinline__ void wait_asynccnt0() {
    asm volatile("s_wait_asynccnt 0x0" ::: "memory");
}

// ---------------------------------------------------------------------------
// fp32 -> bf16 elementwise convert
// ---------------------------------------------------------------------------
__global__ void cvt_f32_bf16(const float* __restrict__ in,
                             unsigned short* __restrict__ out, int n) {
    int i = blockIdx.x * blockDim.x + threadIdx.x;
    if (i < n) out[i] = f2bf(in[i]);
}

// ---------------------------------------------------------------------------
// W[K,N] fp32 -> Wt[N,K] bf16 (transpose so B-fragments are K-contiguous)
// ---------------------------------------------------------------------------
__global__ void cvt_transpose(const float* __restrict__ W,
                              unsigned short* __restrict__ Wt, int K, int N) {
    int i = blockIdx.x * blockDim.x + threadIdx.x;
    if (i < K * N) {
        int k = i / N;
        int n = i - k * N;
        Wt[(size_t)n * K + k] = f2bf(W[i]);
    }
}

// ---------------------------------------------------------------------------
// Tiled bf16 WMMA GEMM:  C[M,N] = act(A[M,K] @ Wt[N,K]^T + bias)
//   block tile 128x128x32, 8 waves (4 along M x 2 along N),
//   wave tile 32x64 = 2x4 accumulators of v_wmma_f32_16x16x32_bf16.
//   Double-buffered LDS fed by GLOBAL_LOAD_ASYNC_TO_LDS_B128 (ASYNCcnt):
//   copy of tile k+1 overlaps WMMA compute on tile k; one barrier per k-tile.
// All of M,N divisible by 128 and K by 32 -> no bounds checks, EXEC all ones.
// ---------------------------------------------------------------------------
#define BM 128
#define BN 128
#define BK 32
#define LDSS 40   // padded LDS row stride in bf16 elements (80B: 16B-aligned, conflict-free)

__global__ void __launch_bounds__(256)
gemm_bf16_wmma(const unsigned short* __restrict__ A,   // [M,K] bf16 row-major
               const unsigned short* __restrict__ Wt,  // [N,K] bf16 row-major
               const float* __restrict__ bias,         // [N]
               unsigned short* __restrict__ C,         // [M,N] bf16 row-major
               int M, int N, int K, int relu)
{
    __shared__ __align__(16) unsigned short As[2][BM * LDSS];
    __shared__ __align__(16) unsigned short Bs[2][BN * LDSS];

    const int tid   = threadIdx.x;
    const int lane  = tid & 31;
    const int wave  = tid >> 5;
    const int lrow  = lane & 15;     // M (or N) index within 16x16 tile
    const int half  = lane >> 4;     // lane half selects K-group per ISA layout
    const int waveM = wave & 3;      // 4 waves along M
    const int waveN = wave >> 2;     // 2 waves along N
    const int blockM = blockIdx.y * BM;
    const int blockN = blockIdx.x * BN;

    v8f acc[2][4] = {};

    // --- async staging addresses -------------------------------------------
    const int ldRow  = tid >> 2;        // 0..63 (two passes cover 128 rows)
    const int ldQuad = (tid & 3) * 8;   // bf16 element offset: 0,8,16,24 (16B chunks)

    const unsigned short* Abase = A  + (size_t)blockM * K;   // uniform -> SGPRs
    const unsigned short* Bbase = Wt + (size_t)blockN * K;   // uniform -> SGPRs
    const unsigned vBase  = (unsigned)((ldRow * K + ldQuad) * 2);   // per-lane byte off
    const unsigned vPass  = (unsigned)(64 * K * 2);                 // +64 rows
    const unsigned ldsPassBytes = 64u * LDSS * 2u;

    unsigned ldsA[2], ldsB[2];
    ldsA[0] = lds_addr32(&As[0][ldRow * LDSS + ldQuad]);
    ldsA[1] = lds_addr32(&As[1][ldRow * LDSS + ldQuad]);
    ldsB[0] = lds_addr32(&Bs[0][ldRow * LDSS + ldQuad]);
    ldsB[1] = lds_addr32(&Bs[1][ldRow * LDSS + ldQuad]);

    auto issue_copy = [&](int buf, int kt) {
        const unsigned ktB = (unsigned)(kt * 2);
        #pragma unroll
        for (int p = 0; p < 2; ++p) {
            async_load_b128(ldsA[buf] + p * ldsPassBytes, vBase + p * vPass + ktB, Abase);
            async_load_b128(ldsB[buf] + p * ldsPassBytes, vBase + p * vPass + ktB, Bbase);
        }
    };

    // --- fragment LDS addresses --------------------------------------------
    // Per ISA 16-bit A-matrix layout: lane holds row (lrow); two contiguous
    // 8-element K runs at k = half*8 and k = 16 + half*8 -> 2x ds_load_b128.
    union Frag { uint4v u[2]; v16bf v; };

    // --- pipeline: copy(k+1) overlaps compute(k) ---------------------------
    const int T = K / BK;
    issue_copy(0, 0);
    wait_asynccnt0();
    __syncthreads();

    for (int kt = 0; kt < T; ++kt) {
        const int cur = kt & 1;
        if (kt + 1 < T) issue_copy(cur ^ 1, (kt + 1) * BK);

        Frag fa[2], fb[4];
        #pragma unroll
        for (int mi = 0; mi < 2; ++mi) {
            const int r = waveM * 32 + mi * 16 + lrow;
            fa[mi].u[0] = *(const uint4v*)(&As[cur][r * LDSS + half * 8]);
            fa[mi].u[1] = *(const uint4v*)(&As[cur][r * LDSS + 16 + half * 8]);
        }
        #pragma unroll
        for (int ni = 0; ni < 4; ++ni) {
            const int c = waveN * 64 + ni * 16 + lrow;
            fb[ni].u[0] = *(const uint4v*)(&Bs[cur][c * LDSS + half * 8]);
            fb[ni].u[1] = *(const uint4v*)(&Bs[cur][c * LDSS + 16 + half * 8]);
        }

        #pragma unroll
        for (int mi = 0; mi < 2; ++mi)
            #pragma unroll
            for (int ni = 0; ni < 4; ++ni)
                acc[mi][ni] = __builtin_amdgcn_wmma_f32_16x16x32_bf16(
                    /*neg_a=*/false, fa[mi].v, /*neg_b=*/false, fb[ni].v,
                    /*c_mod=*/(short)0, acc[mi][ni],
                    /*reuse_a=*/false, /*reuse_b=*/false);

        wait_asynccnt0();   // next buffer's copy complete (this thread's ops)
        __syncthreads();    // cross-thread visibility + buffer-reuse safety
    }

    // Epilogue: acc VGPR r -> (row = r + half*8, col = lrow) within the 16x16 tile
    #pragma unroll
    for (int mi = 0; mi < 2; ++mi) {
        #pragma unroll
        for (int ni = 0; ni < 4; ++ni) {
            const int col = blockN + waveN * 64 + ni * 16 + lrow;
            const float b = bias[col];
            #pragma unroll
            for (int r = 0; r < 8; ++r) {
                const int row = blockM + waveM * 32 + mi * 16 + half * 8 + r;
                float v = acc[mi][ni][r] + b;
                if (relu) v = fmaxf(v, 0.0f);
                C[(size_t)row * N + col] = f2bf(v);
            }
        }
    }
}

// ---------------------------------------------------------------------------
// rec_loss = sum((X - rec)^2)
// ---------------------------------------------------------------------------
__global__ void rec_loss_kernel(const float* __restrict__ X,
                                const unsigned short* __restrict__ rec,
                                float* __restrict__ out, int n) {
    __shared__ float red[256];
    float s = 0.0f;
    for (int i = blockIdx.x * blockDim.x + threadIdx.x; i < n;
         i += gridDim.x * blockDim.x) {
        float d = X[i] - bf2f(rec[i]);
        s += d * d;
    }
    red[threadIdx.x] = s;
    __syncthreads();
    for (int o = 128; o > 0; o >>= 1) {
        if (threadIdx.x < (unsigned)o) red[threadIdx.x] += red[threadIdx.x + o];
        __syncthreads();
    }
    if (threadIdx.x == 0) atomicAdd(out, red[0]);
}

// ---------------------------------------------------------------------------
// dist_loss = 0.5 * sum((latent - clusters[cid])^2), latent is [16384,128]
// ---------------------------------------------------------------------------
__global__ void dist_loss_kernel(const unsigned short* __restrict__ latent,
                                 const float* __restrict__ clusters,
                                 const int* __restrict__ cid,
                                 float* __restrict__ out, int n) {
    __shared__ float red[256];
    float s = 0.0f;
    for (int i = blockIdx.x * blockDim.x + threadIdx.x; i < n;
         i += gridDim.x * blockDim.x) {
        int row = i >> 7;
        int j   = i & 127;
        float d = bf2f(latent[i]) - clusters[cid[row] * 128 + j];
        s += 0.5f * d * d;
    }
    red[threadIdx.x] = s;
    __syncthreads();
    for (int o = 128; o > 0; o >>= 1) {
        if (threadIdx.x < (unsigned)o) red[threadIdx.x] += red[threadIdx.x + o];
        __syncthreads();
    }
    if (threadIdx.x == 0) atomicAdd(out, red[0]);
}

// ---------------------------------------------------------------------------
// Host-side orchestration
// ---------------------------------------------------------------------------
extern "C" void kernel_launch(void* const* d_in, const int* in_sizes, int n_in,
                              void* d_out, int out_size, void* d_ws, size_t ws_size,
                              hipStream_t stream) {
    (void)in_sizes; (void)n_in; (void)out_size; (void)ws_size;

    const int M = 16384;
    // layer l: K = dims[l], N = dims[l+1]
    const int dims[9] = {1024, 1024, 1024, 4096, 128, 4096, 1024, 1024, 1024};

    // Input order: X, cluster_id, enc_W[0..3], enc_b[0..3], dec_W[0..3], dec_b[0..3], clusters
    const float* X        = (const float*)d_in[0];
    const int*   cid      = (const int*)d_in[1];
    const float* Wf[8];
    const float* bf[8];
    for (int l = 0; l < 4; ++l) { Wf[l]     = (const float*)d_in[2 + l];  bf[l]     = (const float*)d_in[6 + l]; }
    for (int l = 0; l < 4; ++l) { Wf[4 + l] = (const float*)d_in[10 + l]; bf[4 + l] = (const float*)d_in[14 + l]; }
    const float* clusters = (const float*)d_in[18];

    // Workspace carve-out (bf16 buffers)
    char*  ws  = (char*)d_ws;
    size_t off = 0;
    auto carve = [&](size_t bytes) -> void* {
        void* p = ws + off;
        off = (off + bytes + 255) & ~(size_t)255;
        return p;
    };
    unsigned short* Xbf = (unsigned short*)carve((size_t)M * 1024 * 2);
    unsigned short* Wt[8];
    for (int l = 0; l < 8; ++l)
        Wt[l] = (unsigned short*)carve((size_t)dims[l] * dims[l + 1] * 2);
    unsigned short* actA   = (unsigned short*)carve((size_t)M * 4096 * 2);
    unsigned short* actB   = (unsigned short*)carve((size_t)M * 4096 * 2);
    unsigned short* latent = (unsigned short*)carve((size_t)M * 128 * 2);

    hipMemsetAsync(d_out, 0, sizeof(float), stream);

    // Convert X and (transpose-)convert all weights to bf16
    {
        int n = M * 1024;
        cvt_f32_bf16<<<(n + 255) / 256, 256, 0, stream>>>(X, Xbf, n);
    }
    for (int l = 0; l < 8; ++l) {
        int n = dims[l] * dims[l + 1];
        cvt_transpose<<<(n + 255) / 256, 256, 0, stream>>>(Wf[l], Wt[l], dims[l], dims[l + 1]);
    }

    auto gemm = [&](const unsigned short* Ain, int l, unsigned short* Cout, int relu) {
        const int K = dims[l], N = dims[l + 1];
        dim3 grid(N / BN, M / BM);
        gemm_bf16_wmma<<<grid, 256, 0, stream>>>(Ain, Wt[l], bf[l], Cout, M, N, K, relu);
    };

    // Encoder
    gemm(Xbf,  0, actA,   1);   // 1024 -> 1024, relu
    gemm(actA, 1, actB,   1);   // 1024 -> 1024, relu
    gemm(actB, 2, actA,   1);   // 1024 -> 4096, relu
    gemm(actA, 3, latent, 0);   // 4096 -> 128, linear (latent)
    // Decoder
    gemm(latent, 4, actA, 1);   // 128  -> 4096, relu
    gemm(actA,   5, actB, 1);   // 4096 -> 1024, relu
    gemm(actB,   6, actA, 1);   // 1024 -> 1024, relu
    gemm(actA,   7, actB, 0);   // 1024 -> 1024, linear (rec)

    // Losses
    rec_loss_kernel<<<1024, 256, 0, stream>>>(X, actB, (float*)d_out, M * 1024);
    dist_loss_kernel<<<512, 256, 0, stream>>>(latent, clusters, cid, (float*)d_out, M * 128);
}
